// Self_Attn_18665927868632
// MI455X (gfx1250) — matrix operand order repaired and checked by reading, not compiled, not run on hardware
//
#include <hip/hip_runtime.h>

#define IN_DIM 256
#define QK     32
#define NB     4
#define NPOS   4096   // 64*64

typedef __attribute__((ext_vector_type(16))) _Float16 v16h;
typedef __attribute__((ext_vector_type(8)))  float    v8f;

// A-matrix (16x32 f16) per-lane K base for VGPR r, lane-half hf.
// contiguous: K = 2r + 16*hf (lanes 0-15: K=0..15 ; lanes 16-31: K=16..31)
__device__ __forceinline__ int a_kb(int r, int hf) {
  return 2 * r + 16 * hf;
}

// ---------------------------------------------------------------------------
// Stage A: q/k/v projections (1x1 convs == channel GEMMs), f32 in -> f16 out.
// One wave per (batch, 16-wide position block). x B-fragments for all 8
// K-steps are cached in registers and reused for 20 output row blocks.
// q,k stored transposed [pos][32]; v stored channel-major [256][4096].
// ---------------------------------------------------------------------------
__global__ void __launch_bounds__(128)
qkv_proj(const float* __restrict__ x,
         const float* __restrict__ fw, const float* __restrict__ fb,
         const float* __restrict__ gw, const float* __restrict__ gb,
         const float* __restrict__ hw, const float* __restrict__ hb,
         _Float16* __restrict__ qT, _Float16* __restrict__ kT,
         _Float16* __restrict__ vm)
{
  const int wid  = blockIdx.x * 4 + (threadIdx.x >> 5);
  const int lane = threadIdx.x & 31;
  const int n    = lane & 15;   // column / A-row index
  const int hf   = lane >> 4;   // lane half
  const int b    = wid >> 8;
  const int p0   = (wid & 255) * 16;

  const float* xb = x + (size_t)b * IN_DIM * NPOS;

  // cache all 8 B fragments (K = 256 -> 8 steps of 32)
  v16h bf[8];
#pragma unroll
  for (int ks = 0; ks < 8; ++ks) {
#pragma unroll
    for (int r = 0; r < 8; ++r) {
      const int c = ks * 32 + 2 * r + 16 * hf;     // B layout: K = 2r + 16*hf
      bf[ks][2 * r]     = (_Float16)xb[(size_t)c * NPOS + p0 + n];
      bf[ks][2 * r + 1] = (_Float16)xb[(size_t)(c + 1) * NPOS + p0 + n];
    }
  }

  // ---- q (2 row blocks) ----
#pragma unroll
  for (int ob = 0; ob < 2; ++ob) {
    v8f acc;
#pragma unroll
    for (int r = 0; r < 8; ++r) acc[r] = fb[ob * 16 + r + 8 * hf];
#pragma unroll
    for (int ks = 0; ks < 8; ++ks) {
      v16h af;
      const float* wrow = fw + (size_t)(ob * 16 + n) * IN_DIM;
#pragma unroll
      for (int r = 0; r < 8; ++r) {
        const int k = ks * 32 + a_kb(r, hf);
        af[2 * r] = (_Float16)wrow[k];  af[2 * r + 1] = (_Float16)wrow[k + 1];
      }
      acc = __builtin_amdgcn_wmma_f32_16x16x32_f16(false, af, false, bf[ks],
                                                   (short)0, acc, false, false);
    }
#pragma unroll
    for (int r = 0; r < 8; ++r)
      qT[((size_t)b * NPOS + p0 + n) * QK + ob * 16 + r + 8 * hf] = (_Float16)acc[r];
  }

  // ---- k (2 row blocks) ----
#pragma unroll
  for (int ob = 0; ob < 2; ++ob) {
    v8f acc;
#pragma unroll
    for (int r = 0; r < 8; ++r) acc[r] = gb[ob * 16 + r + 8 * hf];
#pragma unroll
    for (int ks = 0; ks < 8; ++ks) {
      v16h af;
      const float* wrow = gw + (size_t)(ob * 16 + n) * IN_DIM;
#pragma unroll
      for (int r = 0; r < 8; ++r) {
        const int k = ks * 32 + a_kb(r, hf);
        af[2 * r] = (_Float16)wrow[k];  af[2 * r + 1] = (_Float16)wrow[k + 1];
      }
      acc = __builtin_amdgcn_wmma_f32_16x16x32_f16(false, af, false, bf[ks],
                                                   (short)0, acc, false, false);
    }
#pragma unroll
    for (int r = 0; r < 8; ++r)
      kT[((size_t)b * NPOS + p0 + n) * QK + ob * 16 + r + 8 * hf] = (_Float16)acc[r];
  }

  // ---- v (16 row blocks, channel-major output) ----
#pragma unroll
  for (int ob = 0; ob < 16; ++ob) {
    v8f acc;
#pragma unroll
    for (int r = 0; r < 8; ++r) acc[r] = hb[ob * 16 + r + 8 * hf];
#pragma unroll
    for (int ks = 0; ks < 8; ++ks) {
      v16h af;
      const float* wrow = hw + (size_t)(ob * 16 + n) * IN_DIM;
#pragma unroll
      for (int r = 0; r < 8; ++r) {
        const int k = ks * 32 + a_kb(r, hf);
        af[2 * r] = (_Float16)wrow[k];  af[2 * r + 1] = (_Float16)wrow[k + 1];
      }
      acc = __builtin_amdgcn_wmma_f32_16x16x32_f16(false, af, false, bf[ks],
                                                   (short)0, acc, false, false);
    }
#pragma unroll
    for (int r = 0; r < 8; ++r)
      vm[((size_t)b * IN_DIM + ob * 16 + r + 8 * hf) * NPOS + p0 + n] = (_Float16)acc[r];
  }
}

// ---------------------------------------------------------------------------
// Stage B: per source row j, online row-max m_j and Z_j = sum_i exp(S-m_j).
// S[j,i] = q_j . k_i ; K = 32 -> one WMMA per 16x16 tile.
// One wave per (batch, 16-row j block).
// ---------------------------------------------------------------------------
__global__ void __launch_bounds__(128)
softmax_stats(const _Float16* __restrict__ qT, const _Float16* __restrict__ kT,
              float* __restrict__ mOut, float* __restrict__ zOut)
{
  const int wid  = blockIdx.x * 4 + (threadIdx.x >> 5);
  const int lane = threadIdx.x & 31;
  const int n    = lane & 15;
  const int hf   = lane >> 4;
  const int b    = wid >> 8;
  const int j0   = (wid & 255) * 16;

  v16h aq;
  {
    const _Float16* row = qT + ((size_t)b * NPOS + j0 + n) * QK;
#pragma unroll
    for (int r = 0; r < 8; ++r) {
      const int k = a_kb(r, hf);
      aq[2 * r] = row[k];  aq[2 * r + 1] = row[k + 1];
    }
  }

  float m[8], z[8];
#pragma unroll
  for (int r = 0; r < 8; ++r) { m[r] = -1e30f; z[r] = 0.0f; }

  for (int i0 = 0; i0 < NPOS; i0 += 16) {
    v16h bk;
    const _Float16* row = kT + ((size_t)b * NPOS + i0 + n) * QK;
#pragma unroll
    for (int r = 0; r < 8; ++r) {
      bk[2 * r] = row[2 * r + 16 * hf];  bk[2 * r + 1] = row[2 * r + 1 + 16 * hf];
    }
    v8f zero = {};
    v8f s = __builtin_amdgcn_wmma_f32_16x16x32_f16(false, aq, false, bk,
                                                   (short)0, zero, false, false);
#pragma unroll
    for (int r = 0; r < 8; ++r) {
      float v  = s[r];
      float rm = v;                       // row-max across the 16-lane half
      rm = fmaxf(rm, __shfl_xor(rm, 1, 32));
      rm = fmaxf(rm, __shfl_xor(rm, 2, 32));
      rm = fmaxf(rm, __shfl_xor(rm, 4, 32));
      rm = fmaxf(rm, __shfl_xor(rm, 8, 32));
      const float mn = fmaxf(m[r], rm);
      float p = __expf(v - mn);
      p += __shfl_xor(p, 1, 32);
      p += __shfl_xor(p, 2, 32);
      p += __shfl_xor(p, 4, 32);
      p += __shfl_xor(p, 8, 32);
      z[r] = z[r] * __expf(m[r] - mn) + p;
      m[r] = mn;
    }
  }
  if (n == 0) {                           // lanes 0 and 16 own disjoint rows
#pragma unroll
    for (int r = 0; r < 8; ++r) {
      mOut[(size_t)b * NPOS + j0 + r + 8 * hf] = m[r];
      zOut[(size_t)b * NPOS + j0 + r + 8 * hf] = z[r];
    }
  }
}

// ---------------------------------------------------------------------------
// Stage C: o[c,i] = sum_j v[c,j] * exp(q_j.k_i - m_j)/Z_j ; out = gamma*o + x.
// One wave per (batch, 16-col i block, half of the 256 channels).
// P tile (32x16) staged through per-wave LDS column-major for the B operand.
// ---------------------------------------------------------------------------
__global__ void __launch_bounds__(128)
attn_out(const float* __restrict__ x, const float* __restrict__ gamma,
         const _Float16* __restrict__ qT, const _Float16* __restrict__ kT,
         const _Float16* __restrict__ vm,
         const float* __restrict__ mBuf, const float* __restrict__ zBuf,
         float* __restrict__ out)
{
  __shared__ _Float16 lds[4][16 * 32];    // per-wave 32x16 P tile, col-major
  const int warp = threadIdx.x >> 5;
  const int wid  = blockIdx.x * 4 + warp;
  const int lane = threadIdx.x & 31;
  const int n    = lane & 15;
  const int hf   = lane >> 4;
  const int b    = wid >> 9;              // / 512
  const int rem  = wid & 511;
  const int i0   = (rem >> 1) * 16;
  const int c0   = (rem & 1) * 128;
  _Float16* myl  = lds[warp];

  // fixed B fragment of k for this i block
  v16h bk;
  {
    const _Float16* row = kT + ((size_t)b * NPOS + i0 + n) * QK;
#pragma unroll
    for (int r = 0; r < 8; ++r) {
      bk[2 * r] = row[2 * r + 16 * hf];  bk[2 * r + 1] = row[2 * r + 1 + 16 * hf];
    }
  }

  v8f acc[8];
#pragma unroll
  for (int cb = 0; cb < 8; ++cb) { v8f zz = {}; acc[cb] = zz; }

  const float g = gamma[0];
  const float* mB = mBuf + (size_t)b * NPOS;
  const float* zB = zBuf + (size_t)b * NPOS;

  for (int j0 = 0; j0 < NPOS; j0 += 32) {
    // S tiles for j0..j0+15 and j0+16..j0+31
    v16h a0, a1;
    {
      const _Float16* r0 = qT + ((size_t)b * NPOS + j0 + n) * QK;
      const _Float16* r1 = qT + ((size_t)b * NPOS + j0 + 16 + n) * QK;
#pragma unroll
      for (int r = 0; r < 8; ++r) {
        const int k = a_kb(r, hf);
        a0[2 * r] = r0[k];  a0[2 * r + 1] = r0[k + 1];
        a1[2 * r] = r1[k];  a1[2 * r + 1] = r1[k + 1];
      }
    }
    v8f zero = {};
    v8f s0 = __builtin_amdgcn_wmma_f32_16x16x32_f16(false, a0, false, bk,
                                                    (short)0, zero, false, false);
    v8f s1 = __builtin_amdgcn_wmma_f32_16x16x32_f16(false, a1, false, bk,
                                                    (short)0, zero, false, false);

    __syncthreads();                       // prior reads done before overwrite
#pragma unroll
    for (int r = 0; r < 8; ++r) {
      const int row0 = j0 + r + 8 * hf;
      const int row1 = row0 + 16;
      const float p0 = __expf(s0[r] - mB[row0]) / zB[row0];
      const float p1 = __expf(s1[r] - mB[row1]) / zB[row1];
      myl[n * 32 + r + 8 * hf]      = (_Float16)p0;   // col-major [i][j]
      myl[n * 32 + 16 + r + 8 * hf] = (_Float16)p1;
    }
    __syncthreads();

    v16h bp;                               // P as B operand (K = j local)
#pragma unroll
    for (int r = 0; r < 8; ++r) {
      bp[2 * r]     = myl[n * 32 + 2 * r + 16 * hf];
      bp[2 * r + 1] = myl[n * 32 + 2 * r + 1 + 16 * hf];
    }

#pragma unroll
    for (int cb = 0; cb < 8; ++cb) {       // V(16x32) x P(32x16) accumulate
      v16h av;
      const _Float16* vr = vm + ((size_t)b * IN_DIM + c0 + cb * 16 + n) * NPOS + j0;
#pragma unroll
      for (int r = 0; r < 8; ++r) {
        const int k = a_kb(r, hf);
        av[2 * r] = vr[k];  av[2 * r + 1] = vr[k + 1];
      }
      acc[cb] = __builtin_amdgcn_wmma_f32_16x16x32_f16(false, av, false, bp,
                                                       (short)0, acc[cb], false, false);
    }
  }

  // epilogue: out = gamma * o + x
#pragma unroll
  for (int cb = 0; cb < 8; ++cb) {
#pragma unroll
    for (int r = 0; r < 8; ++r) {
      const int c = c0 + cb * 16 + r + 8 * hf;
      const size_t idx = ((size_t)b * IN_DIM + c) * NPOS + i0 + n;
      out[idx] = g * acc[cb][r] + x[idx];
    }
  }
}

// ---------------------------------------------------------------------------
extern "C" void kernel_launch(void* const* d_in, const int* in_sizes, int n_in,
                              void* d_out, int out_size, void* d_ws, size_t ws_size,
                              hipStream_t stream) {
  const float* x     = (const float*)d_in[0];
  const float* fw    = (const float*)d_in[1];
  const float* fb    = (const float*)d_in[2];
  const float* gw    = (const float*)d_in[3];
  const float* gb    = (const float*)d_in[4];
  const float* hw    = (const float*)d_in[5];
  const float* hb    = (const float*)d_in[6];
  const float* gamma = (const float*)d_in[7];
  float* out = (float*)d_out;

  // workspace layout (all f16 GEMM operands live in L2: ~10.6 MB total)
  _Float16* qT = (_Float16*)d_ws;                         // [B][4096][32]
  _Float16* kT = qT + (size_t)NB * NPOS * QK;             // [B][4096][32]
  _Float16* vm = kT + (size_t)NB * NPOS * QK;             // [B][256][4096]
  float* mBuf  = (float*)(vm + (size_t)NB * IN_DIM * NPOS);
  float* zBuf  = mBuf + (size_t)NB * NPOS;

  // 1024 waves: (B=4) x (4096/16 position blocks)
  qkv_proj<<<256, 128, 0, stream>>>(x, fw, fb, gw, gb, hw, hb, qT, kT, vm);
  // 1024 waves: (B=4) x (4096/16 j blocks)
  softmax_stats<<<256, 128, 0, stream>>>(qT, kT, mBuf, zBuf);
  // 2048 waves: (B=4) x (4096/16 i blocks) x (2 channel halves)
  attn_out<<<512, 128, 0, stream>>>(x, gamma, qT, kT, vm, mBuf, zBuf, out);
}
